// RefineUIGraphLayer_90718299226372
// MI455X (gfx1250) — matrix operand order repaired and checked by reading, not compiled, not run on hardware
//
#include <hip/hip_runtime.h>
#include <hip/hip_bf16.h>

// ---------------------------------------------------------------------------
// Fused RefineUIGraphLayer for MI455X (gfx1250, wave32, WMMA bf16)
//   S = U @ SF^T  (4096 x 65536, K=64)      [GEMM1, v_wmma_f32_16x16x32_bf16]
//   rowAbs[n] = sum_m |S[n,m]|              [fused]
//   AGG^T = SF^T @ S^T                       [GEMM2, v_wmma_f32_16x16x32_bf16]
//   out = U + (AGG/(max(rowAbs,eps)*M)) @ W^T + b
// D->B repack between the GEMMs done with v_permlane16_swap (no LDS/DS ops).
// ---------------------------------------------------------------------------

typedef __bf16 bf16_t;
typedef bf16_t v16bf __attribute__((ext_vector_type(16)));
typedef bf16_t v8bf  __attribute__((ext_vector_type(8)));
typedef bf16_t v2bf  __attribute__((ext_vector_type(2)));
typedef float  v8f   __attribute__((ext_vector_type(8)));
typedef unsigned v8u __attribute__((ext_vector_type(8)));

#define EMB    64
#define MTOT   65536
#define NUSR   4096
#define MSPLIT 8       // M-slabs across workgroups
#define WAVES  8       // waves per workgroup (256 threads, wave32)
#define UT_WG  32      // users per workgroup (2 WMMA tiles of 16)

static __device__ __forceinline__ v8bf ld8(const bf16_t* p) {
  return *reinterpret_cast<const v8bf*>(p);   // 16B aligned at every call site
}
static __device__ __forceinline__ v16bf cat16(v8bf a, v8bf b) {
  return __builtin_shufflevector(a, b, 0,1,2,3,4,5,6,7,8,9,10,11,12,13,14,15);
}
static __device__ __forceinline__ v8f zero8() {
  v8f z;
#pragma unroll
  for (int i = 0; i < 8; ++i) z[i] = 0.f;
  return z;
}
static __device__ __forceinline__ v8f wmma_bf16(v16bf a, v16bf b, v8f c) {
  // 8 args: (neg_a, A, neg_b, B, c_mod, C, reuse_a, reuse_b)
  return __builtin_amdgcn_wmma_f32_16x16x32_bf16(false, a, false, b,
                                                 (short)0, c, false, false);
}
static __device__ __forceinline__ unsigned pack_bf16(float a, float b) {
  v2bf q; q[0] = (bf16_t)a; q[1] = (bf16_t)b;
  return __builtin_bit_cast(unsigned, q);
}

// Exchange half-wave data between two dwords:
//   on return: x = {x.lanes0-15, y.lanes0-15}, y = {x.lanes16-31, y.lanes16-31}
static __device__ __forceinline__ void half_swap(unsigned& x, unsigned& y,
                                                 int hi) {
#if __has_builtin(__builtin_amdgcn_permlane16_swap)
  (void)hi;
  auto r = __builtin_amdgcn_permlane16_swap(x, y, false, false);
  x = r[0]; y = r[1];
#elif __has_builtin(__builtin_amdgcn_permlanex16)
  // identity cross-half copy: lane i <- lane i^16
  unsigned xs = __builtin_amdgcn_permlanex16(x, x, 0x76543210u, 0xfedcba98u,
                                             false, false);
  unsigned ys = __builtin_amdgcn_permlanex16(y, y, 0x76543210u, 0xfedcba98u,
                                             false, false);
  unsigned nx = hi ? ys : x;   // lanes>=16 take y's low half
  unsigned ny = hi ? y  : xs;  // lanes<16  take x's high half
  x = nx; y = ny;
#else
  unsigned xs = (unsigned)__shfl_xor((int)x, 16, 32);
  unsigned ys = (unsigned)__shfl_xor((int)y, 16, 32);
  unsigned nx = hi ? ys : x;
  unsigned ny = hi ? y  : xs;
  x = nx; y = ny;
#endif
}

// --- Kernel 1: f32 -> bf16 copies of U and SF, bf16 SF^T, zero accumulators -
__global__ __launch_bounds__(256) void prep_kernel(
    const float* __restrict__ uf, const float* __restrict__ sf,
    bf16_t* __restrict__ ub, bf16_t* __restrict__ sfb,
    bf16_t* __restrict__ sftb, float* __restrict__ agg,
    float* __restrict__ rowabs)
{
  size_t i = (size_t)blockIdx.x * 256 + threadIdx.x;
  if (i < (size_t)MTOT * EMB) {
    bf16_t h = (bf16_t)sf[i];
    sfb[i] = h;
    size_t m = i >> 6, d = i & 63;
    sftb[d * (size_t)MTOT + m] = h;          // transposed copy for GEMM2 A
  }
  if (i < (size_t)NUSR * EMB) { ub[i] = (bf16_t)uf[i]; agg[i] = 0.f; }
  if (i < (size_t)NUSR) rowabs[i] = 0.f;
}

// --- Kernel 2: fused GEMM1 + |.| row-sum + GEMM2 (partial accumulation) -----
__global__ __launch_bounds__(256) void fused_gemm_kernel(
    const bf16_t* __restrict__ sfb, const bf16_t* __restrict__ sftb,
    const bf16_t* __restrict__ ub, float* __restrict__ agg,
    float* __restrict__ rowabs)
{
  __shared__ float aggLds[UT_WG * EMB];      // 8 KB cross-wave reduction buffer
  __shared__ float absLds[UT_WG];

  const int utile = blockIdx.x / MSPLIT;     // 0..127 (32-user tile)
  const int slab  = blockIdx.x % MSPLIT;     // 0..7   (M slab)
  const int wv    = threadIdx.x >> 5;        // wave in WG
  const int lane  = threadIdx.x & 31;
  const int lo    = lane & 15;
  const int hi    = lane >> 4;               // lane-half selector

  for (int i = threadIdx.x; i < UT_WG * EMB; i += 256) aggLds[i] = 0.f;
  if (threadIdx.x < UT_WG) absLds[threadIdx.x] = 0.f;
  __syncthreads();

  const int ubase     = utile * UT_WG;
  const int mPerWG    = MTOT / MSPLIT;       // 8192
  const int mPerWave  = mPerWG / WAVES;      // 1024
  const int mstart    = slab * mPerWG + wv * mPerWave;

  // GEMM1 B-operand: U^T, K = d.  Lane n (0-15) holds K=0..7,16..23, lanes
  // 16-31 hold K=8..15,24..31 (same K order as the A operand).
  v16bf bU[2][2];
#pragma unroll
  for (int ut = 0; ut < 2; ++ut) {
    const bf16_t* up = ub + (size_t)(ubase + ut * 16 + lo) * EMB + hi * 8;
    bU[ut][0] = cat16(ld8(up +  0), ld8(up + 16));   // d = 0..31
    bU[ut][1] = cat16(ld8(up + 32), ld8(up + 48));   // d = 32..63
  }

  v8f acc2[4][2];                            // AGG^T tiles: 4 d-tiles x 2 u-tiles
#pragma unroll
  for (int dt = 0; dt < 4; ++dt)
#pragma unroll
    for (int ut = 0; ut < 2; ++ut) acc2[dt][ut] = zero8();

  float aacc0 = 0.f, aacc1 = 0.f;            // per-user |S| partial sums

  for (int mc = 0; mc < mPerWave; mc += 32) {
    const int mb = mstart + mc;

    // GEMM1 A-operand: SF rows (lane-half K chunks per A layout)
    const bf16_t* a0 = sfb + (size_t)(mb + lo)      * EMB;
    const bf16_t* a1 = sfb + (size_t)(mb + 16 + lo) * EMB;
    if (mc + 32 < mPerWave) {                 // speculative L2 prefetch
      __builtin_prefetch(a0 + 32 * EMB, 0, 1);
      __builtin_prefetch(a1 + 32 * EMB, 0, 1);
    }
    v16bf A0k0 = cat16(ld8(a0 +      hi * 8), ld8(a0 + 16 + hi * 8));
    v16bf A0k1 = cat16(ld8(a0 + 32 + hi * 8), ld8(a0 + 48 + hi * 8));
    v16bf A1k0 = cat16(ld8(a1 +      hi * 8), ld8(a1 + 16 + hi * 8));
    v16bf A1k1 = cat16(ld8(a1 + 32 + hi * 8), ld8(a1 + 48 + hi * 8));

    v16bf B2[2];
#pragma unroll
    for (int ut = 0; ut < 2; ++ut) {
      // T = SF_tile x U^T  (D: lane = user column, rows j / j+8 per half)
      v8f c0 = zero8(), c1 = zero8();
      c0 = wmma_bf16(A0k0, bU[ut][0], c0);   // m tile 0, d 0..31
      c0 = wmma_bf16(A0k1, bU[ut][1], c0);   // m tile 0, d 32..63
      c1 = wmma_bf16(A1k0, bU[ut][0], c1);   // m tile 1
      c1 = wmma_bf16(A1k1, bU[ut][1], c1);

      // L1 partial: lane (lo,hi) covers user u = ubase+ut*16+lo, 8 m rows each
      float s = 0.f;
#pragma unroll
      for (int j = 0; j < 8; ++j)
        s += __builtin_fabsf(c0[j]) + __builtin_fabsf(c1[j]);
      if (ut == 0) aacc0 += s; else aacc1 += s;

      // Repack D-layout S into GEMM2 B-operand (S^T, K=m 0..31):
      //   pack K pairs to bf16 (B order used here: VGPR v holds K=16h+2v,16h+2v+1), then one
      //   half-wave register exchange per VGPR pair. Pure VALU, no LDS.
      unsigned pc0[4], pc1[4];
#pragma unroll
      for (int v = 0; v < 4; ++v) {
        pc0[v] = pack_bf16(c0[2 * v], c0[2 * v + 1]);
        pc1[v] = pack_bf16(c1[2 * v], c1[2 * v + 1]);
      }
#pragma unroll
      for (int v = 0; v < 4; ++v) half_swap(pc0[v], pc1[v], hi);
      // dwords 0-3 = K 0..7 / 16..23 per half, dwords 4-7 = K 8..15 / 24..31
      v8u t;
#pragma unroll
      for (int v = 0; v < 4; ++v) { t[v] = pc0[v]; t[4 + v] = pc1[v]; }
      B2[ut] = __builtin_bit_cast(v16bf, t);
    }

    // GEMM2: AGG^T += SF^T_tile x S^T ; A from row-major SF^T workspace
#pragma unroll
    for (int dt = 0; dt < 4; ++dt) {
      const bf16_t* ap = sftb + (size_t)(dt * 16 + lo) * MTOT + mb;
      v16bf A2 = cat16(ld8(ap + hi * 16), ld8(ap + hi * 16 + 8));
      acc2[dt][0] = wmma_bf16(A2, B2[0], acc2[dt][0]);
      acc2[dt][1] = wmma_bf16(A2, B2[1], acc2[dt][1]);
    }
  }

  // Cross-wave reduction in LDS (ds_add_f32), then sparse global atomics.
  atomicAdd(&absLds[lo],      aacc0);
  atomicAdd(&absLds[16 + lo], aacc1);
#pragma unroll
  for (int dt = 0; dt < 4; ++dt)
#pragma unroll
    for (int ut = 0; ut < 2; ++ut)
#pragma unroll
      for (int j = 0; j < 8; ++j) {
        int d  = dt * 16 + hi * 8 + j;     // AGG^T row -> d
        int ul = ut * 16 + lo;             // local user
        atomicAdd(&aggLds[ul * EMB + d], acc2[dt][ut][j]);
      }
  __syncthreads();

  for (int i = threadIdx.x; i < UT_WG * EMB; i += 256)
    atomicAdd(&agg[(size_t)ubase * EMB + i], aggLds[i]);
  if (threadIdx.x < UT_WG)
    atomicAdd(&rowabs[ubase + threadIdx.x], absLds[threadIdx.x]);
}

// --- Kernel 3: normalize, @W^T + b, residual add ----------------------------
__global__ __launch_bounds__(256) void finalize_kernel(
    const float* __restrict__ uf, const float* __restrict__ W,
    const float* __restrict__ bias, const float* __restrict__ agg,
    const float* __restrict__ rowabs, float* __restrict__ out)
{
  int gid = blockIdx.x * 256 + threadIdx.x;   // 4096*64 threads
  int n = gid >> 6, d = gid & 63;
  float scale = 1.0f / (fmaxf(rowabs[n], 1e-12f) * (float)MTOT);
  const float* ag = agg + (size_t)n * EMB;
  const float* wr = W   + (size_t)d * EMB;    // W[out,in] row
  float acc = 0.f;
#pragma unroll
  for (int k = 0; k < EMB; ++k) acc = fmaf(ag[k] * scale, wr[k], acc);
  out[gid] = uf[gid] + acc + bias[d];
}

// ---------------------------------------------------------------------------
extern "C" void kernel_launch(void* const* d_in, const int* in_sizes, int n_in,
                              void* d_out, int out_size, void* d_ws, size_t ws_size,
                              hipStream_t stream) {
  const float* uf = (const float*)d_in[0];   // user_feat [4096,64]
  const float* sf = (const float*)d_in[1];   // seq_feat  [2048,32,64] -> [65536,64]
  const float* W  = (const float*)d_in[2];   // [64,64]
  const float* bb = (const float*)d_in[3];   // [64]
  float* out = (float*)d_out;

  char* ws = (char*)d_ws;
  bf16_t* sfb    = (bf16_t*)(ws);                                  // 8 MB
  bf16_t* sftb   = (bf16_t*)(ws + (size_t)MTOT * EMB * 2);         // 8 MB
  bf16_t* ub     = (bf16_t*)(ws + (size_t)MTOT * EMB * 4);         // 512 KB
  float*  agg    = (float*) (ws + (size_t)MTOT * EMB * 4
                                + (size_t)NUSR * EMB * 2);         // 1 MB
  float*  rowabs = (float*) ((char*)agg + (size_t)NUSR * EMB * 4); // 16 KB

  prep_kernel<<<(MTOT * EMB) / 256, 256, 0, stream>>>(uf, sf, ub, sfb, sftb,
                                                      agg, rowabs);
  fused_gemm_kernel<<<(NUSR / UT_WG) * MSPLIT, 256, 0, stream>>>(sfb, sftb, ub,
                                                                 agg, rowabs);
  finalize_kernel<<<(NUSR * EMB) / 256, 256, 0, stream>>>(uf, W, bb, agg,
                                                          rowabs, out);
}
